// MoEAttentionBlock_83305185673378
// MI455X (gfx1250) — matrix operand order repaired and checked by reading, not compiled, not run on hardware
//
#include <hip/hip_runtime.h>
#include <hip/hip_bf16.h>

typedef __attribute__((ext_vector_type(16))) _Float16 v16h;
typedef __attribute__((ext_vector_type(8)))  _Float16 v8h;
typedef __attribute__((ext_vector_type(4)))  _Float16 v4h;
typedef __attribute__((ext_vector_type(8)))  float    v8f;
typedef __attribute__((ext_vector_type(4)))  float    v4f;

union V16 { v16h v; v8h h[2]; };

#define B_  4
#define N_  512
#define H_  1024
#define NH_ 16
#define E_  8
#define DH_ 64
#define T_  (B_ * N_)
#define H3_ (3 * H_)

// ---- CDNA5 async memory->LDS copy (ASYNCcnt-tracked), 16B per lane --------
__device__ __forceinline__ void async_b128(void* lds_dst, const void* gsrc) {
    unsigned off = (unsigned)(uintptr_t)lds_dst;   // LDS byte offset (addr[31:0])
    asm volatile("global_load_async_to_lds_b128 %0, %1, off"
                 :: "v"(off), "v"(gsrc) : "memory");
}
#define WAIT_ASYNC(n) asm volatile("s_wait_asynccnt " #n ::: "memory")

// ---------------------------------------------------------------------------
// Zero expert counters
// ---------------------------------------------------------------------------
__global__ void init_counts_kernel(int* counts, int* cursors) {
    int i = threadIdx.x;
    if (i < E_) { counts[i] = 0; cursors[i] = 0; }
}

// ---------------------------------------------------------------------------
// Gating: one wave per token; top-1 softmax prob + expert index.
// ---------------------------------------------------------------------------
__global__ __launch_bounds__(256)
void gate_kernel(const float* __restrict__ x, const float* __restrict__ wg,
                 float* __restrict__ prob, int* __restrict__ eidx,
                 int* __restrict__ counts) {
    int lane = threadIdx.x & 31;
    int wid  = threadIdx.x >> 5;
    int t = blockIdx.x * 8 + wid;
    if (t >= T_) return;

    float acc[E_];
#pragma unroll
    for (int e = 0; e < E_; ++e) acc[e] = 0.f;

    const float* xr = x + (size_t)t * H_;
    for (int h = lane; h < H_; h += 32) {
        float xv = xr[h];
        const float* wr = wg + (size_t)h * E_;
#pragma unroll
        for (int e = 0; e < E_; ++e) acc[e] += xv * wr[e];
    }
#pragma unroll
    for (int e = 0; e < E_; ++e) {
#pragma unroll
        for (int off = 16; off > 0; off >>= 1)
            acc[e] += __shfl_xor(acc[e], off, 32);
    }
    if (lane == 0) {
        float m = acc[0]; int best = 0;
#pragma unroll
        for (int e = 1; e < E_; ++e)
            if (acc[e] > m) { m = acc[e]; best = e; }
        float s = 0.f;
#pragma unroll
        for (int e = 0; e < E_; ++e) s += __expf(acc[e] - m);
        prob[t] = 1.f / s;
        eidx[t] = best;
        atomicAdd(&counts[best], 1);
    }
}

__global__ void scan_kernel(const int* __restrict__ counts, int* __restrict__ offsets) {
    if (threadIdx.x == 0) {
        int run = 0;
        for (int e = 0; e < E_; ++e) { offsets[e] = run; run += counts[e]; }
    }
}

__global__ __launch_bounds__(256)
void scatter_kernel(const int* __restrict__ eidx, const int* __restrict__ offsets,
                    int* __restrict__ cursors, int* __restrict__ list) {
    int t = blockIdx.x * blockDim.x + threadIdx.x;
    if (t >= T_) return;
    int e = eidx[t];
    int pos = atomicAdd(&cursors[e], 1);
    list[offsets[e] + pos] = t;
}

// ---------------------------------------------------------------------------
// f32 -> f16, 8 elements / thread
// ---------------------------------------------------------------------------
__global__ __launch_bounds__(256)
void convert_x_kernel(const float* __restrict__ x, _Float16* __restrict__ xh, int n8) {
    int i = blockIdx.x * blockDim.x + threadIdx.x;
    if (i >= n8) return;
    const float* src = x + (size_t)i * 8;
    v8h v;
#pragma unroll
    for (int j = 0; j < 8; ++j) v[j] = (_Float16)src[j];
    *(v8h*)(xh + (size_t)i * 8) = v;
}

// ---------------------------------------------------------------------------
// One-time weight transform: W f32 [E][K][Nout] -> WT f16 [E][Nout][K].
// LDS-tiled 64x64 transpose: coalesced f32 reads, b128 f16 writes.
// Halves steady-state GEMM weight traffic and makes B tiles linear copies.
// ---------------------------------------------------------------------------
__global__ __launch_bounds__(256)
void wtrans_kernel(const float* __restrict__ W, _Float16* __restrict__ WT,
                   int K, int Nout) {
    __shared__ _Float16 tile[64][64 + 8];   // [n][k], 144B rows (16B aligned)
    int e  = blockIdx.z;
    int k0 = blockIdx.y * 64;
    int n0 = blockIdx.x * 64;
    const float* We  = W  + (size_t)e * K * Nout;
    _Float16*    WTe = WT + (size_t)e * K * Nout;
    int tid = threadIdx.x;
#pragma unroll
    for (int it = 0; it < 4; ++it) {
        int i  = tid + it * 256;           // 1024 chunks of 4 floats
        int r  = i >> 4;                   // k row
        int c4 = (i & 15) * 4;             // n col
        v4f wv = *(const v4f*)&We[(size_t)(k0 + r) * Nout + n0 + c4];
        tile[c4 + 0][r] = (_Float16)wv[0];
        tile[c4 + 1][r] = (_Float16)wv[1];
        tile[c4 + 2][r] = (_Float16)wv[2];
        tile[c4 + 3][r] = (_Float16)wv[3];
    }
    __syncthreads();
#pragma unroll
    for (int it = 0; it < 2; ++it) {
        int i  = tid + it * 256;           // 512 chunks of 8 halfs
        int n  = i >> 3;
        int kc = (i & 7) * 8;
        *(v8h*)&WTe[(size_t)(n0 + n) * K + k0 + kc] = *(const v8h*)&tile[n][kc];
    }
}

// ---------------------------------------------------------------------------
// Grouped MoE GEMM with async double-buffered LDS staging.
// Block tile 64(M) x 128(N), BK=32; 8 waves as 2x4, each 32x32 via 4 WMMAs.
// Per tile per wave: 3 async_b128 issues (1 A + 2 B) -> WAIT_ASYNC(3)
// waits exactly the previous tile while the next one is in flight.
// ---------------------------------------------------------------------------
__global__ __launch_bounds__(256)
void moe_gemm_kernel(const _Float16* __restrict__ X, const _Float16* __restrict__ WT,
                     const float* __restrict__ bias, const float* __restrict__ prob,
                     const int* __restrict__ offsets, const int* __restrict__ counts,
                     const int* __restrict__ list, float* __restrict__ Out,
                     int K, int Nout) {
    constexpr int BM = 64, BN = 128, BK = 32;
    constexpr int LD = BK + 8;                 // 40 halfs = 80B rows (16B aligned)
    __shared__ _Float16 As[2][BM][LD];         // [m][k]
    __shared__ _Float16 Bs[2][BN][LD];         // [n][k]  (WT is already [n][k])
    __shared__ int toks[BM];

    int e   = blockIdx.z;
    int cnt = counts[e];
    int m0  = blockIdx.y * BM;
    if (m0 >= cnt) return;
    int n0  = blockIdx.x * BN;
    int tid = threadIdx.x;

    if (tid < BM) {
        int r = m0 + tid;
        toks[tid] = (r < cnt) ? list[offsets[e] + r] : -1;
    }
    __syncthreads();

    const _Float16* WTe = WT + (size_t)e * K * Nout;   // [Nout][K]
    int lane = tid & 31, wid = tid >> 5;
    int wm = (wid >> 2) * 32;
    int wn = (wid & 3) * 32;
    int ml = lane & 15;
    int kh = lane >> 4;

    // A staging: 64 rows x 4 chunks = 256 -> 1/thread. Invalid rows read a
    // clamped (valid) address; their outputs are masked at store time, and
    // row m of D depends only on row m of A, so garbage never escapes.
    int am  = tid >> 2;
    int ach = (tid & 3) * 8;
    int atk = toks[am];
    const _Float16* aptr = X + (size_t)(atk >= 0 ? atk : 0) * K;

    auto issue_tile = [&](int k0, int buf) {
        async_b128(&As[buf][am][ach], aptr + k0 + ach);
#pragma unroll
        for (int it = 0; it < 2; ++it) {
            int i  = tid + it * 256;               // 512 chunks
            int bn = i >> 2, bch = (i & 3) * 8;
            async_b128(&Bs[buf][bn][bch], WTe + (size_t)(n0 + bn) * K + k0 + bch);
        }
    };

    v8f acc00 = {}, acc01 = {}, acc10 = {}, acc11 = {};
    int nT = K / BK;
    issue_tile(0, 0);
    for (int t = 0; t < nT; ++t) {
        int buf = t & 1;
        if (t + 1 < nT) { issue_tile((t + 1) * BK, buf ^ 1); WAIT_ASYNC(3); }
        else            { WAIT_ASYNC(0); }
        __syncthreads();

        V16 a0, a1, b0, b1;
        a0.h[0] = *(const v8h*)&As[buf][wm + ml][kh * 8];
        a0.h[1] = *(const v8h*)&As[buf][wm + ml][16 + kh * 8];
        a1.h[0] = *(const v8h*)&As[buf][wm + 16 + ml][kh * 8];
        a1.h[1] = *(const v8h*)&As[buf][wm + 16 + ml][16 + kh * 8];
        b0.h[0] = *(const v8h*)&Bs[buf][wn + ml][kh * 8];
        b0.h[1] = *(const v8h*)&Bs[buf][wn + ml][16 + kh * 8];
        b1.h[0] = *(const v8h*)&Bs[buf][wn + 16 + ml][kh * 8];
        b1.h[1] = *(const v8h*)&Bs[buf][wn + 16 + ml][16 + kh * 8];

        acc00 = __builtin_amdgcn_wmma_f32_16x16x32_f16(false, a0.v, false, b0.v, (short)0, acc00, false, false);
        acc01 = __builtin_amdgcn_wmma_f32_16x16x32_f16(false, a0.v, false, b1.v, (short)0, acc01, false, false);
        acc10 = __builtin_amdgcn_wmma_f32_16x16x32_f16(false, a1.v, false, b0.v, (short)0, acc10, false, false);
        acc11 = __builtin_amdgcn_wmma_f32_16x16x32_f16(false, a1.v, false, b1.v, (short)0, acc11, false, false);
        __syncthreads();
    }

#pragma unroll
    for (int i = 0; i < 8; ++i) {
        int r0 = wm + i + 8 * kh;
        int r1 = r0 + 16;
        int t0 = toks[r0], t1 = toks[r1];
        int nA = n0 + wn + ml, nB = nA + 16;
        if (t0 >= 0) {
            float p = prob[t0];
            Out[(size_t)t0 * Nout + nA] = p * (acc00[i] + bias[e * Nout + nA]);
            Out[(size_t)t0 * Nout + nB] = p * (acc01[i] + bias[e * Nout + nB]);
        }
        if (t1 >= 0) {
            float p = prob[t1];
            Out[(size_t)t1 * Nout + nA] = p * (acc10[i] + bias[e * Nout + nA]);
            Out[(size_t)t1 * Nout + nB] = p * (acc11[i] + bias[e * Nout + nB]);
        }
    }
}

// ---------------------------------------------------------------------------
// Split qkv [T,3H] f32 into per-head Q/K/V [B*NH, N, DH] f16 and also
// V transposed [B*NH, DH, N] f16 (so PV staging is a linear async copy).
// ---------------------------------------------------------------------------
__global__ __launch_bounds__(256)
void split_heads_kernel(const float* __restrict__ qkv, _Float16* __restrict__ qh,
                        _Float16* __restrict__ kh, _Float16* __restrict__ vh,
                        _Float16* __restrict__ vt) {
    int idx = blockIdx.x * blockDim.x + threadIdx.x;   // over T_*H_/8
    if (idx >= T_ * H_ / 8) return;
    int t   = idx >> 7;
    int rem = idx & 127;
    int h   = rem >> 3;
    int d8  = (rem & 7) * 8;
    int b   = t / N_, n = t % N_;
    int bh  = b * NH_ + h;
    size_t dst = ((size_t)bh * N_ + n) * DH_ + d8;
    const float* row = qkv + (size_t)t * H3_ + h * DH_ + d8;
    v8h q, k, v;
#pragma unroll
    for (int j = 0; j < 8; ++j) {
        q[j] = (_Float16)row[j];
        k[j] = (_Float16)row[H_ + j];
        v[j] = (_Float16)row[2 * H_ + j];
    }
    *(v8h*)(qh + dst) = q;
    *(v8h*)(kh + dst) = k;
    *(v8h*)(vh + dst) = v;
#pragma unroll
    for (int j = 0; j < 8; ++j)
        vt[((size_t)bh * DH_ + d8 + j) * N_ + n] = v[j];
}

// ---------------------------------------------------------------------------
// Scores: S = Q K^T / 8 + mask_bias -> probs buffer (pre-softmax).
// Async double-buffered K chunks; K's natural [key][dh] layout IS
// column-major K^T, so B fragments are K-contiguous ds_load_b128.
// Per K chunk: 2 async issues; prologue adds 1 for Q.
// ---------------------------------------------------------------------------
__global__ __launch_bounds__(256)
void scores_kernel(const _Float16* __restrict__ qh, const _Float16* __restrict__ kh,
                   const int* __restrict__ attn_mask, const int* __restrict__ kpm,
                   float* __restrict__ probs) {
    __shared__ _Float16 Qs[32][DH_ + 8];
    __shared__ _Float16 Ks[2][64][DH_ + 8];

    int bh = blockIdx.x;
    int b  = bh / NH_;
    int q0 = blockIdx.y * 32;
    int tid = threadIdx.x;
    int lane = tid & 31, wid = tid >> 5;
    int wm  = (wid >> 2) * 16;
    int wnl = (wid & 3) * 16;
    int ml  = lane & 15, khf = lane >> 4;

    {   // async-stage Q tile (1 issue)
        int m = tid >> 3, d8 = (tid & 7) * 8;
        async_b128(&Qs[m][d8], &qh[((size_t)bh * N_ + q0 + m) * DH_ + d8]);
    }
    auto issue_k = [&](int kt0, int buf) {
#pragma unroll
        for (int it = 0; it < 2; ++it) {
            int i = tid + it * 256;
            int kk = i >> 3, d8 = (i & 7) * 8;
            async_b128(&Ks[buf][kk][d8], &kh[((size_t)bh * N_ + kt0 + kk) * DH_ + d8]);
        }
    };
    issue_k(0, 0);

    int nC = N_ / 64;
    for (int c = 0; c < nC; ++c) {
        int buf = c & 1;
        if (c + 1 < nC) { issue_k((c + 1) * 64, buf ^ 1); WAIT_ASYNC(2); }
        else            { WAIT_ASYNC(0); }
        __syncthreads();

        v8f acc = {};
#pragma unroll
        for (int d0 = 0; d0 < DH_; d0 += 32) {
            V16 a, bb;
            a.h[0]  = *(const v8h*)&Qs[wm + ml][d0 + khf * 8];
            a.h[1]  = *(const v8h*)&Qs[wm + ml][d0 + 16 + khf * 8];
            bb.h[0] = *(const v8h*)&Ks[buf][wnl + ml][d0 + khf * 8];
            bb.h[1] = *(const v8h*)&Ks[buf][wnl + ml][d0 + 16 + khf * 8];
            acc = __builtin_amdgcn_wmma_f32_16x16x32_f16(false, a.v, false, bb.v,
                                                         (short)0, acc, false, false);
        }
        int kt0 = c * 64;
#pragma unroll
        for (int i = 0; i < 8; ++i) {
            int qrow = q0 + wm + i + 8 * khf;
            int kcol = kt0 + wnl + ml;
            int masked = attn_mask[qrow * N_ + kcol] | kpm[b * N_ + kcol];
            float biasv = masked ? -10000.f : 0.f;
            probs[((size_t)bh * N_ + qrow) * N_ + kcol] = acc[i] * 0.125f + biasv;
        }
        __syncthreads();
    }
}

// ---------------------------------------------------------------------------
// Row softmax over 512, one wave per row, in place.
// ---------------------------------------------------------------------------
__global__ __launch_bounds__(256)
void softmax_kernel(float* __restrict__ probs) {
    int lane = threadIdx.x & 31;
    int wid  = threadIdx.x >> 5;
    int row  = blockIdx.x * 8 + wid;
    if (row >= B_ * NH_ * N_) return;
    float* r = probs + (size_t)row * N_;

    float m = -3.4e38f;
    for (int j = lane; j < N_; j += 32) m = fmaxf(m, r[j]);
#pragma unroll
    for (int off = 16; off > 0; off >>= 1) m = fmaxf(m, __shfl_xor(m, off, 32));
    float s = 0.f;
    for (int j = lane; j < N_; j += 32) s += __expf(r[j] - m);
#pragma unroll
    for (int off = 16; off > 0; off >>= 1) s += __shfl_xor(s, off, 32);
    float inv = 1.f / s;
    for (int j = lane; j < N_; j += 32) r[j] = __expf(r[j] - m) * inv;
}

// ---------------------------------------------------------------------------
// ctx = probs @ V.  V tile comes from the pre-transposed [bh][dh][key]
// layout via async copy (1 issue/tile, double-buffered); P is converted
// f32->f16 manually each tile.
// ---------------------------------------------------------------------------
__global__ __launch_bounds__(256)
void pv_kernel(const float* __restrict__ probs, const _Float16* __restrict__ vt,
               _Float16* __restrict__ ctx) {
    __shared__ _Float16 Ps[32][32 + 8];        // [q][k]
    __shared__ _Float16 Vt[2][DH_][32 + 8];    // [dh][key]

    int bh = blockIdx.x;
    int q0 = blockIdx.y * 32;
    int tid = threadIdx.x;
    int lane = tid & 31, wid = tid >> 5;
    int wm  = (wid >> 2) * 16;
    int wnl = (wid & 3) * 16;
    int ml  = lane & 15, khf = lane >> 4;

    auto issue_v = [&](int k0, int buf) {      // 256 chunks -> 1/thread
        int d = tid >> 2, kc = (tid & 3) * 8;
        async_b128(&Vt[buf][d][kc], &vt[((size_t)bh * DH_ + d) * N_ + k0 + kc]);
    };
    issue_v(0, 0);

    v8f acc = {};
    int nT = N_ / 32;
    for (int t = 0; t < nT; ++t) {
        int buf = t & 1;
        {   // stage P (previous iteration's end barrier protects this buffer)
            int m = tid >> 3, c4 = (tid & 7) * 4;
            v4f pv = *(const v4f*)&probs[((size_t)bh * N_ + q0 + m) * N_ + t * 32 + c4];
            v4h ph; ph[0] = (_Float16)pv[0]; ph[1] = (_Float16)pv[1];
            ph[2] = (_Float16)pv[2]; ph[3] = (_Float16)pv[3];
            *(v4h*)&Ps[m][c4] = ph;
        }
        if (t + 1 < nT) { issue_v((t + 1) * 32, buf ^ 1); WAIT_ASYNC(1); }
        else            { WAIT_ASYNC(0); }
        __syncthreads();

        V16 a, bb;
        a.h[0]  = *(const v8h*)&Ps[wm + ml][khf * 8];
        a.h[1]  = *(const v8h*)&Ps[wm + ml][16 + khf * 8];
        bb.h[0] = *(const v8h*)&Vt[buf][wnl + ml][khf * 8];
        bb.h[1] = *(const v8h*)&Vt[buf][wnl + ml][16 + khf * 8];
        acc = __builtin_amdgcn_wmma_f32_16x16x32_f16(false, a.v, false, bb.v,
                                                     (short)0, acc, false, false);
        __syncthreads();
    }

    int b = bh / NH_, h = bh % NH_;
#pragma unroll
    for (int i = 0; i < 8; ++i) {
        int qrow = q0 + wm + i + 8 * khf;
        int t = b * N_ + qrow;
        int d = wnl + ml;
        ctx[(size_t)t * H_ + h * DH_ + d] = (_Float16)acc[i];
    }
}

// ---------------------------------------------------------------------------
// Host-side orchestration
// ---------------------------------------------------------------------------
static inline char* align_up(char* p, size_t a) {
    return (char*)(((uintptr_t)p + a - 1) & ~(uintptr_t)(a - 1));
}

extern "C" void kernel_launch(void* const* d_in, const int* in_sizes, int n_in,
                              void* d_out, int out_size, void* d_ws, size_t ws_size,
                              hipStream_t stream) {
    const float* hs        = (const float*)d_in[0];
    const int*   attn_mask = (const int*)  d_in[1];
    const int*   kpm       = (const int*)  d_in[2];
    const float* w_gate    = (const float*)d_in[3];
    const float* w_qkv     = (const float*)d_in[4];
    const float* b_qkv     = (const float*)d_in[5];
    const float* w_dense   = (const float*)d_in[6];
    const float* b_dense   = (const float*)d_in[7];

    float* out   = (float*)d_out;                  // [B,N,H]
    float* probs = out + (size_t)T_ * H_;          // [B,NH,N,N]

    char* p = (char*)d_ws;
    float* prob   = (float*)p;       p += T_ * sizeof(float);
    int*   eidx   = (int*)p;         p += T_ * sizeof(int);
    int*   counts = (int*)p;         p += E_ * sizeof(int);
    int*   offs   = (int*)p;         p += E_ * sizeof(int);
    int*   curs   = (int*)p;         p += E_ * sizeof(int);
    int*   list   = (int*)p;         p += T_ * sizeof(int);
    p = align_up(p, 256);
    _Float16* xh     = (_Float16*)p; p += (size_t)T_ * H_ * sizeof(_Float16);
    p = align_up(p, 256);
    float* qkv       = (float*)p;    p += (size_t)T_ * H3_ * sizeof(float);
    p = align_up(p, 256);
    _Float16* qh     = (_Float16*)p; p += (size_t)T_ * H_ * sizeof(_Float16);
    _Float16* kh     = (_Float16*)p; p += (size_t)T_ * H_ * sizeof(_Float16);
    _Float16* vh     = (_Float16*)p; p += (size_t)T_ * H_ * sizeof(_Float16);
    _Float16* vt     = (_Float16*)p; p += (size_t)T_ * H_ * sizeof(_Float16);
    _Float16* ctx    = (_Float16*)p; p += (size_t)T_ * H_ * sizeof(_Float16);
    p = align_up(p, 256);
    _Float16* wqkvT  = (_Float16*)p; p += (size_t)E_ * H_ * H3_ * sizeof(_Float16);
    _Float16* wdenT  = (_Float16*)p; p += (size_t)E_ * H_ * H_  * sizeof(_Float16);
    (void)ws_size; (void)n_in; (void)in_sizes; (void)out_size;

    // 0) one-time weight transform: f32 [K][N] -> f16 [N][K]
    wtrans_kernel<<<dim3(H3_ / 64, H_ / 64, E_), 256, 0, stream>>>(w_qkv,  wqkvT, H_, H3_);
    wtrans_kernel<<<dim3(H_  / 64, H_ / 64, E_), 256, 0, stream>>>(w_dense, wdenT, H_, H_);

    // 1) routing
    init_counts_kernel<<<1, 64, 0, stream>>>(counts, curs);
    gate_kernel<<<T_ / 8, 256, 0, stream>>>(hs, w_gate, prob, eidx, counts);
    scan_kernel<<<1, 32, 0, stream>>>(counts, offs);
    scatter_kernel<<<T_ / 256, 256, 0, stream>>>(eidx, offs, curs, list);

    // 2) activations to f16
    convert_x_kernel<<<(T_ * H_ / 8) / 256, 256, 0, stream>>>(hs, xh, T_ * H_ / 8);

    // 3) grouped MoE QKV GEMM (async double-buffered WMMA)
    moe_gemm_kernel<<<dim3(H3_ / 128, T_ / 64, E_), 256, 0, stream>>>(
        xh, wqkvT, b_qkv, prob, offs, counts, list, qkv, H_, H3_);

    // 4) head split (+ V transpose)
    split_heads_kernel<<<(T_ * H_ / 8) / 256, 256, 0, stream>>>(qkv, qh, kh, vh, vt);

    // 5) attention scores (WMMA) + mask bias -> probs buffer
    scores_kernel<<<dim3(B_ * NH_, N_ / 32), 256, 0, stream>>>(
        qh, kh, attn_mask, kpm, probs);

    // 6) softmax rows in place
    softmax_kernel<<<(B_ * NH_ * N_) / 8, 256, 0, stream>>>(probs);

    // 7) context = probs @ V (WMMA, async V staging)
    pv_kernel<<<dim3(B_ * NH_, N_ / 32), 256, 0, stream>>>(probs, vt, ctx);

    // 8) grouped MoE dense GEMM -> final output
    moe_gemm_kernel<<<dim3(H_ / 128, T_ / 64, E_), 256, 0, stream>>>(
        ctx, wdenT, b_dense, prob, offs, counts, list, out, H_, H_);
}
